// SurgeCollapseNet_43190191129179
// MI455X (gfx1250) — matrix operand order repaired and checked by reading, not compiled, not run on hardware
//
#include <hip/hip_runtime.h>
#include <hip/hip_bf16.h>
#include <stdint.h>

typedef __attribute__((ext_vector_type(16))) __bf16 v16bf;
typedef __attribute__((ext_vector_type(8)))  float  v8f;
typedef __attribute__((ext_vector_type(4)))  int    v4i;

typedef __attribute__((address_space(1))) v4i as1_v4i;
typedef __attribute__((address_space(3))) v4i as3_v4i;

#define NN        4096
#define HC        256     // H*C
#define HEADS     4
#define CDIM      64
#define GOUTD     64
#define HIDD      256
#define NEG_SLOPE 0.2f
#define BN_EPS    1e-5f

// ---------------- CDNA5 async global->LDS (guarded) ----------------
#if __has_builtin(__builtin_amdgcn_global_load_async_to_lds_b128)
#define HAVE_ASYNC_LDS 1
#else
#define HAVE_ASYNC_LDS 0
#endif

__device__ __forceinline__ void async_copy16(const void* g, void* l) {
#if HAVE_ASYNC_LDS
    __builtin_amdgcn_global_load_async_to_lds_b128(
        (as1_v4i*)(uintptr_t)g, (as3_v4i*)(uint32_t)(uintptr_t)l, 0, 0);
#else
    *(uint4*)l = *(const uint4*)g;
#endif
}
__device__ __forceinline__ void wait_async0() {
#if HAVE_ASYNC_LDS
#if __has_builtin(__builtin_amdgcn_s_wait_asynccnt)
    __builtin_amdgcn_s_wait_asynccnt(0);
#else
    asm volatile("s_wait_asynccnt 0" ::: "memory");
#endif
#endif
}

// ---- wave32 WMMA fragment index maps (CDNA5 ISA 7.12.2) ----
// A (16x32, 16-bit): elem[2v+j] holds K = 16*(v>=4) + 8*half + 2*(v&3)+j
__device__ __forceinline__ int a_kmap(int v, int j, int half) {
    return ((v & 4) << 2) + (half << 3) + ((v & 3) << 1) + j;
}
// swizzled-tile index of element (row,col) of an A-operand matrix [rows x ncols]
__device__ __forceinline__ int aswz_idx(int row, int col, int ncols) {
    int kl   = col & 31;
    int tile = (row >> 4) * (ncols >> 5) + (col >> 5);
    int lane = (row & 15) + (((kl >> 3) & 1) << 4);
    int e    = ((kl >> 4) << 3) + (kl & 7);
    return tile * 512 + lane * 16 + e;
}
// B (32x16, 16-bit): lane = col + 16*half, elem e holds K = 16*(e>=8) + 8*half + (e&7)
__device__ __forceinline__ int bswz_idx(int k, int c, int ncols) {
    int tile = (k >> 5) * (ncols >> 4) + (c >> 4);
    int lane = (c & 15) + (((k >> 3) & 1) << 4);
    return tile * 512 + lane * 16 + (((k >> 4) & 1) << 3) + (k & 7);
}
// C/D (16x16 f32): elem r -> row M = r + 8*half, col N = lane&15

// -------------------- K0: zero BN stat accumulators --------------------
__global__ void k0_init(float* __restrict__ p) { p[threadIdx.x] = 0.0f; }

// ------- KW: f32 weight matrix [K x C] -> bf16 B-fragment tiles --------
__global__ __launch_bounds__(256)
void k_wswz(const float* __restrict__ W, __bf16* __restrict__ Wsw,
            int n, int cshift)
{
    int idx = blockIdx.x * 256 + threadIdx.x;
    if (idx >= n) return;
    int k = idx >> cshift;
    int c = idx & ((1 << cshift) - 1);
    Wsw[bswz_idx(k, c, 1 << cshift)] = (__bf16)W[idx];
}

// ------------- K1: h = x^T @ W_gat -> B-fragment-tiled bf16 ------------
// grid 256 (16-row tiles). block 128 = 4 waves; wave w -> cols [64w,64w+64)
// Double-buffered async-LDS stage of the x^T tile; B frags preloaded so the
// 4 WMMAs drain one 8x global_load_b128 clause.
__global__ __launch_bounds__(128)
void k1_proj(const float* __restrict__ x, const __bf16* __restrict__ Wgsw,
             __bf16* __restrict__ hbsw)
{
    __shared__ __align__(16) float lds_x[2][32 * 16];  // [buf][kk][m]
    const int i0   = blockIdx.x * 16;
    const int tid  = threadIdx.x;
    const int wave = tid >> 5;
    const int lane = tid & 31;
    const int half = lane >> 4;
    const int m    = lane & 15;
    const int col  = lane & 15;
    const int kk   = tid >> 2;      // staging row 0..31
    const int part = tid & 3;       // staging 16B chunk in row

    // prologue: stage first tile into buffer 0
    async_copy16(&x[(size_t)kk * NN + i0 + part * 4],
                 &lds_x[0][kk * 16 + part * 4]);

    v8f acc[4] = {};
    const int STEPS = NN / 32;
    for (int s = 0; s < STEPS; ++s) {
        const int cur = s & 1;
        wait_async0();       // my async copies (into cur) are done
        __syncthreads();     // everyone's copies done; prev reads of nxt done
        if (s + 1 < STEPS) { // prefetch next tile while computing this one
            async_copy16(&x[(size_t)((s + 1) * 32 + kk) * NN + i0 + part * 4],
                         &lds_x[cur ^ 1][kk * 16 + part * 4]);
        }

        // preload all 4 B fragments (one clause of 8x global_load_b128)
        v16bf bf[4];
#pragma unroll
        for (int t = 0; t < 4; ++t)
            bf[t] = *(const v16bf*)&Wgsw[(s * 16 + wave * 4 + t) * 512 + lane * 16];

        v16bf af;
#pragma unroll
        for (int v = 0; v < 8; ++v)
#pragma unroll
            for (int j = 0; j < 2; ++j)
                af[2 * v + j] = (__bf16)lds_x[cur][a_kmap(v, j, half) * 16 + m];

#pragma unroll
        for (int t = 0; t < 4; ++t)
            acc[t] = __builtin_amdgcn_wmma_f32_16x16x32_bf16(false, af, false, bf[t],
                                                             (short)0, acc[t], false, false);
    }
    // store h in B-fragment tiling (consumed as V by attention)
#pragma unroll
    for (int t = 0; t < 4; ++t) {
        int c = (wave * 4 + t) * 16 + col;
#pragma unroll
        for (int r = 0; r < 8; ++r)
            hbsw[bswz_idx(i0 + r + 8 * half, c, HC)] = (__bf16)acc[t][r];
    }
}

// ------------- K2: per-node attention scalars a_src/a_dst --------------
__global__ __launch_bounds__(256)
void k2_attvec(const __bf16* __restrict__ hbsw, const float* __restrict__ att_src,
               const float* __restrict__ att_dst,
               float* __restrict__ asrc, float* __restrict__ adst)
{
    int gid = blockIdx.x * 256 + threadIdx.x;   // N*H threads
    int n = gid >> 2, hd = gid & 3;
    float s = 0.f, d = 0.f;
#pragma unroll 4
    for (int c = 0; c < CDIM; ++c) {
        float hv = (float)hbsw[bswz_idx(n, hd * CDIM + c, HC)];
        s += hv * att_src[hd * CDIM + c];
        d += hv * att_dst[hd * CDIM + c];
    }
    asrc[n * HEADS + hd] = s;
    adst[n * HEADS + hd] = d;
}

// -------- K3: streamed rank-1-logit attention, out = softmax(e) @ h ----
// grid 256 (16-row i tiles). block 128 = 4 waves, wave = head.
// V fragments straight from L2 (B-fragment-tiled h); a_src double-buffered.
__global__ __launch_bounds__(128)
void k3_attn(const __bf16* __restrict__ hbsw, const float* __restrict__ asrc,
             const float* __restrict__ adst, const float* __restrict__ b_gat,
             __bf16* __restrict__ aout_asw)
{
    __shared__ __align__(16) float lds_as[2][32 * HEADS];
    const int i0   = blockIdx.x * 16;
    const int tid  = threadIdx.x;
    const int hd   = tid >> 5;       // head = wave id
    const int lane = tid & 31;
    const int half = lane >> 4;
    const int m    = lane & 15;
    const int col  = lane & 15;

    const float my_adst = adst[(i0 + m) * HEADS + hd];

    if (tid < 32)   // prologue: 512B a_src tile, 32 x 16B async copies
        async_copy16(&asrc[(size_t)tid * 4], &lds_as[0][tid * 4]);

    v8f acc[4] = {};
    float rowsum = 0.f;
    const int STEPS = NN / 32;
    for (int s = 0; s < STEPS; ++s) {
        const int cur = s & 1;
        const int j0  = s * 32;
        wait_async0();
        __syncthreads();
        if (s + 1 < STEPS && tid < 32)
            async_copy16(&asrc[(size_t)(j0 + 32) * HEADS + tid * 4],
                         &lds_as[cur ^ 1][tid * 4]);

        // preload 4 V fragments
        v16bf bf[4];
#pragma unroll
        for (int t = 0; t < 4; ++t)
            bf[t] = *(const v16bf*)&hbsw[(s * 16 + hd * 4 + t) * 512 + lane * 16];

        // P tile: exp(leaky-relu(a_dst_i + a_src_j)), doubled on the diagonal
        v16bf af;
#pragma unroll
        for (int v = 0; v < 8; ++v) {
#pragma unroll
            for (int j = 0; j < 2; ++j) {
                int kkk = a_kmap(v, j, half);
                float e = my_adst + lds_as[cur][kkk * HEADS + hd];
                e = (e >= 0.f) ? e : NEG_SLOPE * e;
                float p = __expf(e);
                if (j0 + kkk == i0 + m) p += p;   // self-loop duplication
                rowsum += p;
                af[2 * v + j] = (__bf16)p;
            }
        }
#pragma unroll
        for (int t = 0; t < 4; ++t)
            acc[t] = __builtin_amdgcn_wmma_f32_16x16x32_bf16(false, af, false, bf[t],
                                                             (short)0, acc[t], false, false);
    }

    float Z = rowsum + __shfl_xor(rowsum, 16);   // softmax denominator per row
    float invz[8];
#pragma unroll
    for (int r = 0; r < 8; ++r)
        invz[r] = 1.0f / __shfl(Z, r + 8 * half);

#pragma unroll
    for (int t = 0; t < 4; ++t) {
        int c = hd * CDIM + t * 16 + col;
        float bias = b_gat[c];
#pragma unroll
        for (int r = 0; r < 8; ++r) {
            float o = acc[t][r] * invz[r] + bias;
            o = (o > 0.f) ? o : 0.f;             // relu
            aout_asw[aswz_idx(i0 + r + 8 * half, c, HC)] = (__bf16)o;
        }
    }
}

// --- K4: g = aout @ W_fc + b_fc ; accumulate per-column sum / sum^2 ----
// grid 256 (16-row tiles). block 128 = 4 waves, wave w -> cols [16w,16w+16)
__global__ __launch_bounds__(128)
void k4_fc_stats(const __bf16* __restrict__ aout_asw, const __bf16* __restrict__ Wfcsw,
                 const float* __restrict__ bfc,
                 float* __restrict__ colsum, float* __restrict__ colsq)
{
    const int i0   = blockIdx.x * 16;
    const int tid  = threadIdx.x;
    const int wave = tid >> 5;
    const int lane = tid & 31;
    const int half = lane >> 4;
    const int col  = lane & 15;
    const int c0   = wave * 16;

    v8f acc = {};
#pragma unroll
    for (int k0 = 0; k0 < HC; k0 += 32) {
        v16bf af  = *(const v16bf*)&aout_asw[((i0 >> 4) * 8 + (k0 >> 5)) * 512 + lane * 16];
        v16bf bfv = *(const v16bf*)&Wfcsw[((k0 >> 5) * 4 + wave) * 512 + lane * 16];
        acc = __builtin_amdgcn_wmma_f32_16x16x32_bf16(false, af, false, bfv,
                                                      (short)0, acc, false, false);
    }
    float bias = bfc[c0 + col];
    float s = 0.f, ss = 0.f;
#pragma unroll
    for (int r = 0; r < 8; ++r) {
        float g = acc[r] + bias;
        s += g; ss += g * g;
    }
    s  += __shfl_xor(s, 16);
    ss += __shfl_xor(ss, 16);
    if (half == 0) {
        atomicAdd(&colsum[c0 + col], s);
        atomicAdd(&colsq[c0 + col],  ss);
    }
}

// ------- K5: BN-collapse (mean of normalized rows == beta) + MLP -------
__global__ __launch_bounds__(256)
void k5_head(const float* __restrict__ colsum, const float* __restrict__ colsq,
             const float* __restrict__ gamma, const float* __restrict__ beta,
             const float* __restrict__ W1, const float* __restrict__ b1,
             const float* __restrict__ W2, const float* __restrict__ b2,
             const float* __restrict__ W3, const float* __restrict__ b3,
             float* __restrict__ out)
{
    __shared__ float v[GOUTD];
    __shared__ float h1[HIDD];
    __shared__ float h2[HIDD];
    const int t = threadIdx.x;
    if (t < GOUTD) {
        const float invN = 1.0f / (float)NN;
        float mu  = colsum[t] * invN;
        float var = colsq[t] * invN - mu * mu;
        float inv = rsqrtf(var + BN_EPS);
        v[t] = gamma[t] * (colsum[t] * invN - mu) * inv + beta[t];  // == beta
    }
    __syncthreads();
    {
        float s = b1[t];
#pragma unroll 4
        for (int c = 0; c < GOUTD; ++c) s += v[c] * W1[c * HIDD + t];
        h1[t] = (s > 0.f) ? s : 0.f;
    }
    __syncthreads();
    {
        float s = b2[t];
#pragma unroll 4
        for (int k = 0; k < HIDD; ++k) s += h1[k] * W2[k * HIDD + t];
        h2[t] = (s > 0.f) ? s : 0.f;
    }
    __syncthreads();
    if (t < 2) {
        float s = b3[t];
        for (int k = 0; k < HIDD; ++k) s += h2[k] * W3[k * 2 + t];
        out[t] = s;
    }
}

extern "C" void kernel_launch(void* const* d_in, const int* in_sizes, int n_in,
                              void* d_out, int out_size, void* d_ws, size_t ws_size,
                              hipStream_t stream) {
    (void)in_sizes; (void)n_in; (void)out_size; (void)ws_size;
    const float* x       = (const float*)d_in[0];
    const float* W_gat   = (const float*)d_in[1];
    const float* att_src = (const float*)d_in[2];
    const float* att_dst = (const float*)d_in[3];
    const float* b_gat   = (const float*)d_in[4];
    const float* W_fc    = (const float*)d_in[5];
    const float* b_fc    = (const float*)d_in[6];
    const float* bn_g    = (const float*)d_in[7];
    const float* bn_b    = (const float*)d_in[8];
    const float* W1 = (const float*)d_in[9];   const float* b1 = (const float*)d_in[10];
    const float* W2 = (const float*)d_in[11];  const float* b2 = (const float*)d_in[12];
    const float* W3 = (const float*)d_in[13];  const float* b3 = (const float*)d_in[14];

    // workspace layout (all 1 KB aligned)
    char*   ws      = (char*)d_ws;
    __bf16* Wgsw    = (__bf16*)ws;                              // 2 MB
    __bf16* Wfcsw   = (__bf16*)(ws + (size_t)2 * 1024 * 1024);  // 32 KB
    __bf16* hbsw    = (__bf16*)(ws + (size_t)2 * 1024 * 1024 + 64 * 1024);        // 2 MB
    float*  asrc    = (float*) (ws + (size_t)4 * 1024 * 1024 + 64 * 1024);        // 64 KB
    float*  adst    = asrc + NN * HEADS;                                           // 64 KB
    __bf16* aoutasw = (__bf16*)(ws + (size_t)4 * 1024 * 1024 + 192 * 1024);       // 2 MB
    float*  colsum  = (float*) (ws + (size_t)6 * 1024 * 1024 + 192 * 1024);
    float*  colsq   = colsum + GOUTD;                            // contiguous 128 floats

    k0_init    <<<1, 128, 0, stream>>>(colsum);
    k_wswz     <<<(NN * HC) / 256, 256, 0, stream>>>(W_gat, Wgsw, NN * HC, 8);
    k_wswz     <<<(HC * GOUTD) / 256, 256, 0, stream>>>(W_fc, Wfcsw, HC * GOUTD, 6);
    k1_proj    <<<NN / 16, 128, 0, stream>>>(x, Wgsw, hbsw);
    k2_attvec  <<<NN * HEADS / 256, 256, 0, stream>>>(hbsw, att_src, att_dst, asrc, adst);
    k3_attn    <<<NN / 16, 128, 0, stream>>>(hbsw, asrc, adst, b_gat, aoutasw);
    k4_fc_stats<<<NN / 16, 128, 0, stream>>>(aoutasw, Wfcsw, b_fc, colsum, colsq);
    k5_head    <<<1, 256, 0, stream>>>(colsum, colsq, bn_g, bn_b,
                                       W1, b1, W2, b2, W3, b3, (float*)d_out);
}
